// VertexFeatureExtractor_34376918237289
// MI455X (gfx1250) — matrix operand order repaired and checked by reading, not compiled, not run on hardware
//
#include <hip/hip_runtime.h>

// Problem constants (from reference)
#define BB    16
#define NN    4096
#define CC    256     // channels = K of GEMM1
#define DD    256     // hidden = N of all GEMMs
#define HH    128
#define WW    128
#define MTOT  (BB*NN)         // 65536 rows
#define LDSW  264             // padded LDS row stride (halfs), 16B-aligned rows

typedef __attribute__((ext_vector_type(16))) _Float16 v16h;
typedef __attribute__((ext_vector_type(8)))  float    v8f;

union Frag16 { v16h v; uint4 q[2]; };

// ---------------------------------------------------------------------------
// Weight transpose + f16 convert:  Wsrc[c][d] (fp32) -> Wt[d][c] (f16)
// ---------------------------------------------------------------------------
__global__ void transpose_w(const float* __restrict__ Wsrc, _Float16* __restrict__ Wt) {
    long i = (long)blockIdx.x * 256 + threadIdx.x;   // 65536 elements
    int c = (int)(i >> 8);
    int d = (int)(i & 255);
    Wt[(long)d * CC + c] = (_Float16)Wsrc[i];
}

// ---------------------------------------------------------------------------
// Bilinear sampling: features (B,C,H,W) fp32, vertices (B,N,2) in pixels.
// Writes A0[point][channel] as f16 (K-contiguous rows for WMMA A).
// One block = 8 points, 256 threads = 256 channels.
// ---------------------------------------------------------------------------
__global__ __launch_bounds__(256) void sample_k(const float* __restrict__ feat,
                                                const float* __restrict__ verts,
                                                _Float16* __restrict__ A0) {
    const int c = threadIdx.x;
    const long p0 = (long)blockIdx.x * 8;
    #pragma unroll
    for (int p = 0; p < 8; ++p) {
        long pt = p0 + p;
        int  b  = (int)(pt >> 12);                  // / 4096
        float vx = verts[pt * 2 + 0];
        float vy = verts[pt * 2 + 1];
        // grid = 2*v/511 - 1 ; ix = (g+1)*0.5*127 = v * 127/511 ; clip [0,127]
        float ix = fminf(fmaxf(vx * (127.0f / 511.0f), 0.0f), 127.0f);
        float iy = fminf(fmaxf(vy * (127.0f / 511.0f), 0.0f), 127.0f);
        float x0f = floorf(ix), y0f = floorf(iy);
        float wx = ix - x0f,    wy = iy - y0f;
        int x0 = (int)x0f, y0 = (int)y0f;
        int x1 = (x0 + 1 > 127) ? 127 : x0 + 1;
        int y1 = (y0 + 1 > 127) ? 127 : y0 + 1;
        const float* fb = feat + (((long)b * CC + c) << 14);   // * H*W
        float v00 = fb[y0 * WW + x0];
        float v01 = fb[y0 * WW + x1];
        float v10 = fb[y1 * WW + x0];
        float v11 = fb[y1 * WW + x1];
        float top = v00 + (v01 - v00) * wx;
        float bot = v10 + (v11 - v10) * wx;
        float val = top + (bot - top) * wy;
        A0[pt * CC + c] = (_Float16)val;
    }
}

// ---------------------------------------------------------------------------
// Fused GEMM: Y = epilogue( X(65536x256,f16) @ W(256x256) + bias )
// Wt is W transposed (N-major, K-contiguous) in f16.
//
// Block: 256 threads = 8 waves. Grid: (MTOT/256, 2).
// Workgroup tile: 256 rows x 128 cols (blockIdx.y picks col half).
// Wave tile: 32 rows x 128 cols = 2 M-tiles x 8 N-tiles.
// Each B fragment feeds 2 WMMAs (A-row blocking), B frags double-buffered in
// registers so ds_load latency overlaps WMMA issue.
// MODE 0: relu -> f16 out.  MODE 1: + posenc(vertices) -> f16 out.  MODE 2: fp32 out.
// ---------------------------------------------------------------------------
template <int MODE>
__global__ __launch_bounds__(256) void gemm256(const _Float16* __restrict__ X,
                                               const _Float16* __restrict__ Wt,
                                               const float*    __restrict__ bias,
                                               const float*    __restrict__ verts,
                                               _Float16*       __restrict__ Yh,
                                               float*          __restrict__ Yf) {
    __shared__ _Float16 lds_w[128 * LDSW];   // 67.6 KB: this block's 128 weight cols

    const int tid   = threadIdx.x;
    const int cbase = blockIdx.y * 128;

    // Stage Wt[cbase+dd][c] (f16, stride 256) -> lds_w[dd*264 + c], 16B vectors.
    #pragma unroll 4
    for (int it = 0; it < 16; ++it) {
        int idx = (it * 256 + tid) * 8;     // half index into 128*256
        int dd = idx >> 8, c = idx & 255;
        uint4 q = *(const uint4*)(Wt + (long)(cbase + dd) * CC + c);
        *(uint4*)(&lds_w[dd * LDSW + c]) = q;
    }
    __syncthreads();

    const int wave = tid >> 5;
    const int lane = tid & 31;
    const int half = lane >> 4;          // 0: lanes 0-15, 1: lanes 16-31
    const int lq   = lane & 15;
    const long rbase = (long)blockIdx.x * 256 + wave * 32;

    v8f acc0[8], acc1[8];
    #pragma unroll
    for (int j = 0; j < 8; ++j) {
        acc0[j] = (v8f){0,0,0,0,0,0,0,0};
        acc1[j] = (v8f){0,0,0,0,0,0,0,0};
    }

    // A fragment base pointers (16-bit A 16x32 striping: lanes 0-15 K[0..7]+[16..23],
    // lanes 16-31 K[8..15]+[24..31]).
    const _Float16* aptr0 = X + (rbase + lq) * CC + half * 8;
    const _Float16* aptr1 = aptr0 + (long)16 * CC;
    const _Float16* bbase = &lds_w[lq * LDSW + half * 8];

    #pragma unroll
    for (int ks = 0; ks < CC; ks += 32) {
        Frag16 a0, a1;
        a0.q[0] = *(const uint4*)(aptr0 + ks);
        a0.q[1] = *(const uint4*)(aptr0 + ks + 16);
        a1.q[0] = *(const uint4*)(aptr1 + ks);
        a1.q[1] = *(const uint4*)(aptr1 + ks + 16);

        const _Float16* bp = bbase + ks;
        Frag16 bc, bn;
        bc.q[0] = *(const uint4*)(bp);
        bc.q[1] = *(const uint4*)(bp + 16);
        #pragma unroll
        for (int j = 0; j < 8; ++j) {
            if (j < 7) {                    // prefetch next N-tile's B fragment
                const _Float16* bq = bp + (j + 1) * 16 * LDSW;
                bn.q[0] = *(const uint4*)(bq);
                bn.q[1] = *(const uint4*)(bq + 16);
            }
            acc0[j] = __builtin_amdgcn_wmma_f32_16x16x32_f16(
                false, a0.v, false, bc.v, (short)0, acc0[j], false, false);
            acc1[j] = __builtin_amdgcn_wmma_f32_16x16x32_f16(
                false, a1.v, false, bc.v, (short)0, acc1[j], false, false);
            bc = bn;
        }
    }

    // Epilogue. C/D layout: lane l -> col = cbase + jt*16 + (l&15);
    // VGPR j of M-tile i -> row = rbase + i*16 + (l>>4)*8 + j.
    float vx[2][8], vy[2][8];
    if (MODE == 1) {
        #pragma unroll
        for (int i = 0; i < 2; ++i)
            #pragma unroll
            for (int j = 0; j < 8; ++j) {
                long r = rbase + i * 16 + half * 8 + j;
                vx[i][j] = verts[r * 2 + 0];
                vy[i][j] = verts[r * 2 + 1];
            }
    }
    #pragma unroll
    for (int t = 0; t < 8; ++t) {
        int col = cbase + t * 16 + lq;
        float bcol = bias[col];
        #pragma unroll
        for (int i = 0; i < 2; ++i) {
            #pragma unroll
            for (int j = 0; j < 8; ++j) {
                long r = rbase + i * 16 + half * 8 + j;
                float v = (i ? acc1[t][j] : acc0[t][j]) + bcol;
                if (MODE == 0) {
                    v = fmaxf(v, 0.0f);
                    Yh[r * DD + col] = (_Float16)v;
                } else if (MODE == 1) {
                    // sinusoidal positional encoding, d_model = 128 per axis
                    int   e    = col & 127;
                    float sel  = (col < 128) ? vx[i][j] : vy[i][j];
                    float freq = __expf(-0.07195578425f * (float)(e & ~1));
                    float ang  = sel * (1000.0f / 512.0f) * freq;
                    float pe   = (e & 1) ? __cosf(ang) : __sinf(ang);
                    Yh[r * DD + col] = (_Float16)(v + pe);
                } else {
                    Yf[r * DD + col] = v;
                }
            }
        }
    }
}

// ---------------------------------------------------------------------------
extern "C" void kernel_launch(void* const* d_in, const int* in_sizes, int n_in,
                              void* d_out, int out_size, void* d_ws, size_t ws_size,
                              hipStream_t stream) {
    const float* features = (const float*)d_in[0];   // (16,256,128,128)
    const float* vertices = (const float*)d_in[1];   // (16,4096,2)
    const float* W1 = (const float*)d_in[3];
    const float* b1 = (const float*)d_in[4];
    const float* W2 = (const float*)d_in[5];
    const float* b2 = (const float*)d_in[6];
    const float* Wp = (const float*)d_in[7];
    const float* bp = (const float*)d_in[8];

    char* ws = (char*)d_ws;
    _Float16* A0  = (_Float16*)ws;                         // 32 MB (reused for Z)
    _Float16* H1  = (_Float16*)(ws + (size_t)(32 << 20));  // 32 MB
    _Float16* W1t = (_Float16*)(ws + (size_t)(64 << 20));  // 128 KB each
    _Float16* W2t = W1t + 65536;
    _Float16* Wpt = W2t + 65536;

    transpose_w<<<256, 256, 0, stream>>>(W1, W1t);
    transpose_w<<<256, 256, 0, stream>>>(W2, W2t);
    transpose_w<<<256, 256, 0, stream>>>(Wp, Wpt);

    sample_k<<<MTOT / 8, 256, 0, stream>>>(features, vertices, A0);

    dim3 ggrid(MTOT / 256, 2);
    // h = relu(A0 @ W1 + b1)
    gemm256<0><<<ggrid, 256, 0, stream>>>(A0, W1t, b1, nullptr, H1, nullptr);
    // z = (h @ W2 + b2) + posenc(vertices)   (stored into A0's buffer)
    gemm256<1><<<ggrid, 256, 0, stream>>>(H1, W2t, b2, vertices, A0, nullptr);
    // out = z @ Wp + bp  (fp32)
    gemm256<2><<<ggrid, 256, 0, stream>>>(A0, Wpt, bp, nullptr, nullptr, (float*)d_out);
}
